// NTXent_34471407518032
// MI455X (gfx1250) — compile-verified, matrix operand order checked by reading, not run on hardware
//
#include <hip/hip_runtime.h>
#include <hip/hip_bf16.h>

// NT-Xent fused loss for MI455X (gfx1250, wave32, WMMA).
// N = 4096, D = 256 per the reference. TEMPERATURE = 0.5 -> 1/T = 2.0.

#define DK        256        // feature dim (K), 8 chunks of 32
#define KCHUNKS   8
#define TEMP_INV  2.0f
#define NSPLIT    8          // column splits of the 2N columns

typedef __attribute__((ext_vector_type(16))) __bf16 v16bf;
typedef __attribute__((ext_vector_type(8)))  float  v8f;

union Frag16 { v16bf v; uint4 q[2]; };

// ---------------------------------------------------------------------------
// Kernel 1: row-normalize z = concat(z_i, z_j), store bf16 (RNE) into d_ws.
// ---------------------------------------------------------------------------
__global__ __launch_bounds__(256)
void ntx_normalize(const float* __restrict__ zi, const float* __restrict__ zj,
                   unsigned short* __restrict__ zn, int N) {
    const int row = blockIdx.x;              // 0 .. 2N-1
    const int t   = threadIdx.x;             // 0 .. 255 == DK
    const float* src = (row < N) ? (zi + (size_t)row * DK)
                                 : (zj + (size_t)(row - N) * DK);
    const float v = src[t];
    float ss = v * v;
    ss += __shfl_xor(ss, 1, 32);
    ss += __shfl_xor(ss, 2, 32);
    ss += __shfl_xor(ss, 4, 32);
    ss += __shfl_xor(ss, 8, 32);
    ss += __shfl_xor(ss, 16, 32);
    __shared__ float wsum[8];
    if ((t & 31) == 0) wsum[t >> 5] = ss;
    __syncthreads();
    float tot = 0.f;
    #pragma unroll
    for (int i = 0; i < 8; ++i) tot += wsum[i];
    const float inv = 1.0f / fmaxf(sqrtf(tot), 1e-8f);
    const float zf = v * inv;
    union { float f; unsigned int u; } cv; cv.f = zf;
    const unsigned int b = cv.u + 0x7FFFu + ((cv.u >> 16) & 1u);  // RNE f32->bf16
    zn[(size_t)row * DK + t] = (unsigned short)(b >> 16);
}

// ---------------------------------------------------------------------------
// Kernel 2: zero the per-row accumulators.
// ---------------------------------------------------------------------------
__global__ void ntx_zero(float* __restrict__ rowsum, float* __restrict__ pos, int n2) {
    const int i = blockIdx.x * blockDim.x + threadIdx.x;
    if (i < n2) { rowsum[i] = 0.0f; pos[i] = 0.0f; }
}

// ---------------------------------------------------------------------------
// Kernel 3: fused Gram-matrix + exp row-sum + positive capture.
// Wave = one 16-row strip; block = 8 waves; grid = (row blocks, NSPLIT).
// K = 256 unrolled as 8x WMMA bf16 16x16x32; B tiles double-buffered.
// Branch-free epilogue: positives accumulated as selects, flushed once.
// ---------------------------------------------------------------------------
__global__ __launch_bounds__(256)
void ntx_simexp(const unsigned short* __restrict__ zn,
                float* __restrict__ rowsum, float* __restrict__ pos,
                int N2, int colsPerSplit) {
    const int lane = threadIdx.x & 31;
    const int wave = threadIdx.x >> 5;
    const int half = lane >> 4;
    const int sub  = lane & 15;
    const int rowBase  = (blockIdx.x * 8 + wave) * 16;
    const int colStart = blockIdx.y * colsPerSplit;
    const int nTiles   = colsPerSplit >> 4;     // 64 (even)
    const int N        = N2 >> 1;

    // Hoisted A fragments: this wave's 16 rows, full K = 256.
    const unsigned short* arow = zn + (size_t)(rowBase + sub) * DK + half * 8;
    Frag16 afrag[KCHUNKS];
    #pragma unroll
    for (int kc = 0; kc < KCHUNKS; ++kc) {
        afrag[kc].q[0] = *(const uint4*)(arow + kc * 32);
        afrag[kc].q[1] = *(const uint4*)(arow + kc * 32 + 16);
    }

    float acc[8], pacc[8];
    #pragma unroll
    for (int g = 0; g < 8; ++g) { acc[g] = 0.0f; pacc[g] = 0.0f; }

    // Per-lane B base pointer for this column split (half*8 folded in).
    const unsigned short* bcol = zn + (size_t)(colStart + sub) * DK + half * 8;
    const size_t tileStride = (size_t)16 * DK;   // elements per 16-column tile

    auto loadB = [&](Frag16* b, const unsigned short* p) {
        #pragma unroll
        for (int kc = 0; kc < KCHUNKS; ++kc) {
            b[kc].q[0] = *(const uint4*)(p + kc * 32);
            b[kc].q[1] = *(const uint4*)(p + kc * 32 + 16);
        }
    };

    auto compute = [&](const Frag16* b, int colBase) {
        v8f c = {};
        #pragma unroll
        for (int kc = 0; kc < KCHUNKS; ++kc) {
            c = __builtin_amdgcn_wmma_f32_16x16x32_bf16(
                    false, afrag[kc].v, false, b[kc].v, (short)0, c, false, false);
        }
        const int cglob = colBase + sub;
        #pragma unroll
        for (int g = 0; g < 8; ++g) {
            const int   r = rowBase + g + 8 * half;
            const float s = c[g];
            const int   partner = (r < N) ? (r + N) : (r - N);
            pacc[g] += (cglob == partner) ? s : 0.0f;                 // positive (once globally)
            acc[g]  += (cglob == r) ? 0.0f : __expf(s * TEMP_INV);    // masked exp
        }
    };

    // Double-buffered column-tile loop (manual 2x unroll; nTiles is even).
    Frag16 b0[KCHUNKS], b1[KCHUNKS];
    loadB(b0, bcol);
    for (int t = 0; t < nTiles; t += 2) {
        loadB(b1, bcol + (size_t)(t + 1) * tileStride);
        compute(b0, colStart + t * 16);
        const int t2 = (t + 2 < nTiles) ? (t + 2) : (nTiles - 1);  // clamped prefetch
        loadB(b0, bcol + (size_t)t2 * tileStride);
        compute(b1, colStart + (t + 1) * 16);
    }

    // Reduce row partials (and positives) across the 16 lanes of each half.
    #pragma unroll
    for (int g = 0; g < 8; ++g) {
        float v = acc[g], p = pacc[g];
        v += __shfl_xor(v, 1, 32);  p += __shfl_xor(p, 1, 32);
        v += __shfl_xor(v, 2, 32);  p += __shfl_xor(p, 2, 32);
        v += __shfl_xor(v, 4, 32);  p += __shfl_xor(p, 4, 32);
        v += __shfl_xor(v, 8, 32);  p += __shfl_xor(p, 8, 32);
        if (sub == 0) {
            const int r = rowBase + g + 8 * half;
            atomicAdd(&rowsum[r], v);
            atomicAdd(&pos[r], p);     // other tiles contribute exact 0.0
        }
    }
}

// ---------------------------------------------------------------------------
// Kernel 4: loss_i = log(rowsum_i) - pos_i/T ; out = sum(w2*loss)/sum(w2)
// ---------------------------------------------------------------------------
__global__ __launch_bounds__(256)
void ntx_finalize(const float* __restrict__ rowsum, const float* __restrict__ pos,
                  const float* __restrict__ weights, float* __restrict__ out, int N2) {
    const int N = N2 >> 1;
    float num = 0.0f, den = 0.0f;
    for (int i = threadIdx.x; i < N2; i += 256) {
        const float w  = weights[(i < N) ? i : (i - N)];
        const float li = logf(rowsum[i]) - pos[i] * TEMP_INV;
        num += w * li;
        den += w;
    }
    num += __shfl_xor(num, 1, 32);  den += __shfl_xor(den, 1, 32);
    num += __shfl_xor(num, 2, 32);  den += __shfl_xor(den, 2, 32);
    num += __shfl_xor(num, 4, 32);  den += __shfl_xor(den, 4, 32);
    num += __shfl_xor(num, 8, 32);  den += __shfl_xor(den, 8, 32);
    num += __shfl_xor(num, 16, 32); den += __shfl_xor(den, 16, 32);
    __shared__ float snum[8], sden[8];
    const int lane = threadIdx.x & 31, wv = threadIdx.x >> 5;
    if (lane == 0) { snum[wv] = num; sden[wv] = den; }
    __syncthreads();
    if (threadIdx.x == 0) {
        float n = 0.0f, d = 0.0f;
        #pragma unroll
        for (int i = 0; i < 8; ++i) { n += snum[i]; d += sden[i]; }
        out[0] = n / d;
    }
}

// ---------------------------------------------------------------------------
extern "C" void kernel_launch(void* const* d_in, const int* in_sizes, int n_in,
                              void* d_out, int out_size, void* d_ws, size_t ws_size,
                              hipStream_t stream) {
    const float* zi = (const float*)d_in[0];
    const float* zj = (const float*)d_in[1];
    const float* w  = (const float*)d_in[2];
    const int N  = in_sizes[2];      // 4096
    const int N2 = 2 * N;            // 8192

    // Workspace: zn bf16 [2N*DK] (4 MB) | rowsum f32 [2N] | pos f32 [2N]
    unsigned short* zn = (unsigned short*)d_ws;
    float* rowsum = (float*)((char*)d_ws + (size_t)N2 * DK * sizeof(unsigned short));
    float* pos    = rowsum + N2;

    ntx_normalize<<<N2, 256, 0, stream>>>(zi, zj, zn, N);
    ntx_zero<<<(N2 + 255) / 256, 256, 0, stream>>>(rowsum, pos, N2);

    dim3 grid(N2 / 128, NSPLIT);     // 64 row-blocks x 8 column splits
    ntx_simexp<<<grid, 256, 0, stream>>>(zn, rowsum, pos, N2, N2 / NSPLIT);

    ntx_finalize<<<1, 256, 0, stream>>>(rowsum, pos, w, (float*)d_out, N2);
}
